// FakeQuantLinear_64415919506229
// MI455X (gfx1250) — 8-bit matrix operand order repaired and checked by reading, compiled, not run on hardware
//
#include <hip/hip_runtime.h>

typedef int v8i __attribute__((ext_vector_type(8)));

#define BM 128
#define BN 128
#define BK 64
#define LDS_PITCH 80

#if __has_builtin(__builtin_amdgcn_global_load_async_to_lds_b128)
#define ASYNC_STAGE 1
typedef int v4i_vec __attribute__((vector_size(16)));
typedef __attribute__((address_space(1))) v4i_vec gv4;
typedef __attribute__((address_space(3))) v4i_vec lv4;
#endif

__global__ __launch_bounds__(256) void fq_quant_x(
    const float* __restrict__ x, const float* __restrict__ scale_p,
    const int* __restrict__ zp_p, signed char* __restrict__ xq,
    int* __restrict__ rowQ, int K)
{
    __shared__ int ssum;
    const int row = blockIdx.x;
    const int t = threadIdx.x;
    if (t == 0) ssum = 0;
    __syncthreads();

    const float sa = *scale_p;
    const float zpf = (float)(*zp_p);
    const float4* xr = (const float4*)(x + (long long)row * K);
    unsigned* outp = (unsigned*)(xq + (long long)row * K);

    int local = 0;
    for (int j = t; j < (K >> 2); j += 256) {
        float4 v = xr[j];
        float r0 = fminf(fmaxf(__builtin_rintf(v.x / sa) + zpf, -128.f), 127.f);
        float r1 = fminf(fmaxf(__builtin_rintf(v.y / sa) + zpf, -128.f), 127.f);
        float r2 = fminf(fmaxf(__builtin_rintf(v.z / sa) + zpf, -128.f), 127.f);
        float r3 = fminf(fmaxf(__builtin_rintf(v.w / sa) + zpf, -128.f), 127.f);
        int q0 = (int)r0, q1 = (int)r1, q2 = (int)r2, q3 = (int)r3;
        local += q0 + q1 + q2 + q3;
        outp[j] = (unsigned)(q0 & 255) | ((unsigned)(q1 & 255) << 8) |
                  ((unsigned)(q2 & 255) << 16) | ((unsigned)(q3 & 255) << 24);
    }
    atomicAdd(&ssum, local);
    __syncthreads();
    if (t == 0) rowQ[row] = ssum;
}

__global__ __launch_bounds__(256) void fq_pack_w(
    const int* __restrict__ w, signed char* __restrict__ wq,
    int* __restrict__ rowW, int K)
{
    __shared__ int ssum;
    const int row = blockIdx.x;
    const int t = threadIdx.x;
    if (t == 0) ssum = 0;
    __syncthreads();

    const int4* wr = (const int4*)(w + (long long)row * K);
    unsigned* outp = (unsigned*)(wq + (long long)row * K);

    int local = 0;
    for (int j = t; j < (K >> 2); j += 256) {
        int4 v = wr[j];
        local += v.x + v.y + v.z + v.w;
        outp[j] = (unsigned)(v.x & 255) | ((unsigned)(v.y & 255) << 8) |
                  ((unsigned)(v.z & 255) << 16) | ((unsigned)(v.w & 255) << 24);
    }
    atomicAdd(&ssum, local);
    __syncthreads();
    if (t == 0) rowW[row] = ssum;
}

#if ASYNC_STAGE
__device__ __forceinline__ void stage_async(const signed char* gbase, int K,
                                            int bRow, int kk, int t,
                                            signed char* lds)
{
#pragma unroll
    for (int i = 0; i < 2; ++i) {
        int idx = t + i * 256;
        int row = idx >> 2, cg = idx & 3;
        const signed char* g =
            gbase + (long long)(bRow + row) * K + kk * BK + cg * 16;
        signed char* l = lds + row * LDS_PITCH + cg * 16;
        __builtin_amdgcn_global_load_async_to_lds_b128(
            (gv4*)g, (lv4*)l, 0, 0);
    }
}
#else
__device__ __forceinline__ void stage_load(const signed char* base, int K,
                                           int bRow, int kk, int t, int4* r)
{
#pragma unroll
    for (int i = 0; i < 2; ++i) {
        int idx = t + i * 256;
        int row = idx >> 2, cg = idx & 3;
        r[i] = *(const int4*)(base + (long long)(bRow + row) * K + kk * BK + cg * 16);
    }
}

__device__ __forceinline__ void stage_store(signed char* lds, int t, const int4* r)
{
#pragma unroll
    for (int i = 0; i < 2; ++i) {
        int idx = t + i * 256;
        int row = idx >> 2, cg = idx & 3;
        *(int4*)(lds + row * LDS_PITCH + cg * 16) = r[i];
    }
}
#endif

__global__ __launch_bounds__(256) void fq_gemm(
    const signed char* __restrict__ Aq,
    const signed char* __restrict__ Wq,
    const int* __restrict__ rowQ, const int* __restrict__ rowW,
    const float* __restrict__ bias,
    const float* __restrict__ sa_p, const int* __restrict__ zpa_p,
    const float* __restrict__ sw_p, const int* __restrict__ wzp_p,
    float* __restrict__ out, int M, int N, int K)
{
    __shared__ __align__(16) signed char As[2][BM * LDS_PITCH];
    __shared__ __align__(16) signed char Bs[2][BN * LDS_PITCH];

    const int t = threadIdx.x;
    const int lane = t & 31;
    const int wave = t >> 5;
    const int waveM = wave >> 2;
    const int waveN = wave & 3;
    const int half = lane >> 4;
    const int l16 = lane & 15;

    const int bM = blockIdx.y * BM;
    const int bN = blockIdx.x * BN;
    const int ksteps = K / BK;

    v8i acc[4][2] = {};

#if ASYNC_STAGE
    stage_async(Aq, K, bM, 0, t, As[0]);
    stage_async(Wq, K, bN, 0, t, Bs[0]);
    asm volatile("s_wait_asynccnt 0x0" ::: "memory");
    __syncthreads();
#else
    int4 ra[2], rb[2];
    stage_load(Aq, K, bM, 0, t, ra);
    stage_load(Wq, K, bN, 0, t, rb);
    stage_store(As[0], t, ra);
    stage_store(Bs[0], t, rb);
    __syncthreads();
#endif

    for (int kk = 0; kk < ksteps; ++kk) {
        const int cur = kk & 1;

#if ASYNC_STAGE
        if (kk + 1 < ksteps) {
            stage_async(Aq, K, bM, kk + 1, t, As[cur ^ 1]);
            stage_async(Wq, K, bN, kk + 1, t, Bs[cur ^ 1]);
        }
#else
        if (kk + 1 < ksteps) {
            stage_load(Aq, K, bM, kk + 1, t, ra);
            stage_load(Wq, K, bN, kk + 1, t, rb);
        }
#endif

        v8i afrag[4];
#pragma unroll
        for (int mt = 0; mt < 4; ++mt) {
            const signed char* ab =
                &As[cur][(waveM * 64 + mt * 16 + l16) * LDS_PITCH];
            int4 alo = *(const int4*)(ab + half * 16);
            int4 ahi = *(const int4*)(ab + 32 + half * 16);
            afrag[mt][0] = alo.x; afrag[mt][1] = alo.y;
            afrag[mt][2] = alo.z; afrag[mt][3] = alo.w;
            afrag[mt][4] = ahi.x; afrag[mt][5] = ahi.y;
            afrag[mt][6] = ahi.z; afrag[mt][7] = ahi.w;
        }

        v8i bfrag[2];
#pragma unroll
        for (int nt = 0; nt < 2; ++nt) {
            const signed char* bb =
                &Bs[cur][(waveN * 32 + nt * 16 + l16) * LDS_PITCH];
            int4 lo = *(const int4*)(bb + half * 16);
            int4 hi = *(const int4*)(bb + 32 + half * 16);
            bfrag[nt][0] = lo.x; bfrag[nt][1] = lo.y;
            bfrag[nt][2] = lo.z; bfrag[nt][3] = lo.w;
            bfrag[nt][4] = hi.x; bfrag[nt][5] = hi.y;
            bfrag[nt][6] = hi.z; bfrag[nt][7] = hi.w;
        }

#pragma unroll
        for (int mt = 0; mt < 4; ++mt)
#pragma unroll
            for (int nt = 0; nt < 2; ++nt)
                acc[mt][nt] = __builtin_amdgcn_wmma_i32_16x16x64_iu8(
                    true, afrag[mt], true, bfrag[nt], acc[mt][nt], false, false);

#if ASYNC_STAGE
        asm volatile("s_wait_asynccnt 0x0" ::: "memory");
        __syncthreads();
#else
        __syncthreads();
        if (kk + 1 < ksteps) {
            stage_store(As[cur ^ 1], t, ra);
            stage_store(Bs[cur ^ 1], t, rb);
        }
        __syncthreads();
#endif
    }

    const float s = (*sa_p) * (*sw_p);
    const int zp = *zpa_p, wzp = *wzp_p;
    const bool corr_on = (zp != 0) || (wzp != 0);

#pragma unroll
    for (int nt = 0; nt < 2; ++nt) {
        const int nG = bN + waveN * 32 + nt * 16 + l16;
        const float bv = bias[nG];
        const int rw = corr_on ? rowW[nG] : 0;
#pragma unroll
        for (int mt = 0; mt < 4; ++mt) {
#pragma unroll
            for (int v = 0; v < 8; ++v) {
                const int mG = bM + waveM * 64 + mt * 16 + half * 8 + v;
                int a = acc[mt][nt][v];
                if (corr_on)
                    a -= zp * rw + wzp * rowQ[mG] - K * zp * wzp;
                out[(long long)mG * N + nG] = s * (float)a + bv;
            }
        }
    }
}

extern "C" void kernel_launch(void* const* d_in, const int* in_sizes, int n_in,
                              void* d_out, int out_size, void* d_ws, size_t ws_size,
                              hipStream_t stream)
{
    const float* x    = (const float*)d_in[0];
    const int*   w    = (const int*)d_in[1];
    const float* bias = (const float*)d_in[2];
    const float* sa   = (const float*)d_in[3];
    const int*   zpa  = (const int*)d_in[4];
    const float* sw   = (const float*)d_in[5];
    const int*   wzp  = (const int*)d_in[6];
    float* out = (float*)d_out;

    const int N = in_sizes[2];
    const int K = in_sizes[1] / N;
    const int M = in_sizes[0] / K;

    signed char* xq = (signed char*)d_ws;
    signed char* wq = xq + (size_t)M * K;
    int* rowQ = (int*)(wq + (size_t)N * K);
    int* rowW = rowQ + M;

    fq_quant_x<<<M, 256, 0, stream>>>(x, sa, zpa, xq, rowQ, K);
    fq_pack_w<<<N, 256, 0, stream>>>(w, wq, rowW, K);

    dim3 grid(N / BN, M / BM);
    fq_gemm<<<grid, 256, 0, stream>>>(xq, wq, rowQ, rowW, bias,
                                      sa, zpa, sw, wzp, out, M, N, K);
}
